// TestAllReduceRMSNormStaticQuantFP8Model_52647709114888
// MI455X (gfx1250) — 8-bit matrix operand order repaired and checked by reading, compiled, not run on hardware
//
#include <hip/hip_runtime.h>
#include <hip/hip_bf16.h>

typedef int   v16i __attribute__((ext_vector_type(16)));
typedef float v8f  __attribute__((ext_vector_type(8)));
typedef int   v4i_vs __attribute__((vector_size(16)));

#define H        1024
#define T_ROWS   32768
#define RMS_EPS  1e-6f
#define FP8_MAX  448.0f

#define AS1 __attribute__((address_space(1)))
#define AS3 __attribute__((address_space(3)))

#if defined(__has_builtin)
# if __has_builtin(__builtin_amdgcn_global_load_async_to_lds_b128) && \
     __has_builtin(__builtin_amdgcn_s_wait_asynccnt)
#  define USE_ASYNC_LDS 1
# endif
#endif
#ifndef USE_ASYNC_LDS
# define USE_ASYNC_LDS 0
#endif

__device__ __forceinline__ unsigned char f32_to_e4m3(float f) {
  unsigned char sign = (__float_as_uint(f) >> 31) ? 0x80 : 0x00;
  if (!(f == f)) return (unsigned char)(sign | 0x7F);
  float a = fabsf(f);
  if (a > FP8_MAX) a = FP8_MAX;
  if (a < 0.015625f) {
    int q = (int)rintf(a * 512.0f);
    if (q >= 8) return (unsigned char)(sign | 0x08);
    return (unsigned char)(sign | q);
  }
  int e; float m = frexpf(a, &e);
  int E = e + 6;
  int q = (int)rintf(m * 16.0f);
  if (q == 16) { q = 8; ++E; }
  if (E > 15) return (unsigned char)(sign | 0x7E);
  return (unsigned char)(sign | (E << 3) | (q & 7));
}

__global__ void quant_weights_kernel(const float* __restrict__ W,
                                     const float* __restrict__ wt_s,
                                     unsigned char* __restrict__ Wq,
                                     int total4) {
  int i = blockIdx.x * blockDim.x + threadIdx.x;
  if (i >= total4) return;
  int layer = (i >> 18);
  float inv = 1.0f / wt_s[layer];
  float4 w = reinterpret_cast<const float4*>(W)[i];
  uchar4 q;
  q.x = f32_to_e4m3(w.x * inv);
  q.y = f32_to_e4m3(w.y * inv);
  q.z = f32_to_e4m3(w.z * inv);
  q.w = f32_to_e4m3(w.w * inv);
  reinterpret_cast<uchar4*>(Wq)[i] = q;
}

__global__ void relu_rms_quant_kernel(const float* __restrict__ x,
                                      const float* __restrict__ norm_w0,
                                      const float* __restrict__ act_s,
                                      float* __restrict__ resid,
                                      unsigned char* __restrict__ yq) {
  const int row = blockIdx.x;
  const int t   = threadIdx.x;
  const int c0  = t * 4;
  const size_t base = (size_t)row * H;

  float4 xv = *reinterpret_cast<const float4*>(x + base + c0);
  float4 r;
  r.x = fmaxf(xv.x, 0.f); r.y = fmaxf(xv.y, 0.f);
  r.z = fmaxf(xv.z, 0.f); r.w = fmaxf(xv.w, 0.f);
  *reinterpret_cast<float4*>(resid + base + c0) = r;

  float ss = r.x * r.x + r.y * r.y + r.z * r.z + r.w * r.w;
  for (int m = 16; m >= 1; m >>= 1) ss += __shfl_xor(ss, m, 32);

  __shared__ float wsum[8];
  __shared__ float sfac;
  if ((t & 31) == 0) wsum[t >> 5] = ss;
  __syncthreads();
  if (t == 0) {
    float tot = 0.f;
    #pragma unroll
    for (int i = 0; i < 8; ++i) tot += wsum[i];
    sfac = rsqrtf(tot * (1.0f / H) + RMS_EPS);
  }
  __syncthreads();

  const float fac   = sfac;
  const float inv_a = 1.0f / act_s[0];
  float4 w = *reinterpret_cast<const float4*>(norm_w0 + c0);
  uchar4 q;
  q.x = f32_to_e4m3(r.x * fac * w.x * inv_a);
  q.y = f32_to_e4m3(r.y * fac * w.y * inv_a);
  q.z = f32_to_e4m3(r.z * fac * w.z * inv_a);
  q.w = f32_to_e4m3(r.w * fac * w.w * inv_a);
  *reinterpret_cast<uchar4*>(yq + base + c0) = q;
}

__global__ void __launch_bounds__(512) fp8_gemm_rms_kernel(
    const unsigned char* __restrict__ Aq,
    const unsigned char* __restrict__ Wq,
    const float* __restrict__ resid_io,
    float* __restrict__ resid_out,
    const float* __restrict__ norm_w,
    const float* __restrict__ act_s,
    const float* __restrict__ wt_s,
    int layer, int last,
    unsigned char* __restrict__ yq_next,
    float* __restrict__ y_out)
{
  __shared__ __align__(16) unsigned char sA[16 * H];
  __shared__ float srow[16];
  __shared__ float smul[16];

  const int tid    = threadIdx.x;
  const int wave   = tid >> 5;
  const int lane   = tid & 31;
  const int lane15 = lane & 15;
  const int hi     = lane >> 4;
  const size_t row0 = (size_t)blockIdx.x * 16;

#if USE_ASYNC_LDS
  {
    AS1 v4i_vs* g = (AS1 v4i_vs*)(uintptr_t)(Aq + row0 * H);
    unsigned lbase = (unsigned)(uintptr_t)&sA[0];
    AS3 v4i_vs* l = (AS3 v4i_vs*)lbase;
    __builtin_amdgcn_global_load_async_to_lds_b128(g + tid,       l + tid,       0, 0);
    __builtin_amdgcn_global_load_async_to_lds_b128(g + tid + 512, l + tid + 512, 0, 0);
    __builtin_amdgcn_s_wait_asynccnt(0);
  }
#else
  {
    const uint4* src = reinterpret_cast<const uint4*>(Aq + row0 * H);
    uint4* dst = reinterpret_cast<uint4*>(sA);
    dst[tid]       = src[tid];
    dst[tid + 512] = src[tid + 512];
  }
#endif
  if (tid < 16) srow[tid] = 0.f;
  __syncthreads();

  const float scale = act_s[layer] * wt_s[layer];

  const int nbase = wave * 64;
  v8f acc[4];
  #pragma unroll
  for (int t = 0; t < 4; ++t)
    #pragma unroll
    for (int r = 0; r < 8; ++r) acc[t][r] = 0.0f;

  #pragma unroll 2
  for (int s = 0; s < 8; ++s) {
    const int k0 = s * 128;
    union { v16i v; uint2 d[8]; } af;
    #pragma unroll
    for (int j = 0; j < 8; ++j)
      af.d[j] = *reinterpret_cast<const uint2*>(&sA[lane15 * H + k0 + (j >> 1) * 32 + hi * 16 + (j & 1) * 8]);

    #pragma unroll
    for (int t = 0; t < 4; ++t) {
      const int n = nbase + t * 16 + lane15;
      const unsigned char* wrow = Wq + (size_t)n * H + k0 + hi * 16;
      union { v16i v; uint4 d[4]; } bf;
      #pragma unroll
      for (int c = 0; c < 4; ++c)
        bf.d[c] = *reinterpret_cast<const uint4*>(wrow + c * 32);
      if (k0 + 128 < H) __builtin_prefetch(wrow + 128, 0, 1);

      acc[t] = __builtin_amdgcn_wmma_f32_16x16x128_fp8_fp8(
                   af.v, bf.v, (short)0, acc[t], false, false);
    }
  }

  float ss[8];
  #pragma unroll
  for (int r = 0; r < 8; ++r) ss[r] = 0.f;

  #pragma unroll
  for (int t = 0; t < 4; ++t) {
    const int n = nbase + t * 16 + lane15;
    #pragma unroll
    for (int r = 0; r < 8; ++r) {
      const int m = r + hi * 8;
      const float old = resid_io[(row0 + m) * H + n];
      const float rn  = acc[t][r] * scale + old;
      acc[t][r] = rn;
      ss[r] += rn * rn;
    }
  }
  #pragma unroll
  for (int r = 0; r < 8; ++r) {
    float v = ss[r];
    v += __shfl_xor(v, 8, 32);
    v += __shfl_xor(v, 4, 32);
    v += __shfl_xor(v, 2, 32);
    v += __shfl_xor(v, 1, 32);
    if (lane15 == 0) atomicAdd(&srow[r + hi * 8], v);
  }
  __syncthreads();
  if (tid < 16) smul[tid] = rsqrtf(srow[tid] * (1.0f / H) + RMS_EPS);
  __syncthreads();

  const float inv_a = last ? 0.f : (1.0f / act_s[layer + 1]);
  #pragma unroll
  for (int t = 0; t < 4; ++t) {
    const int n  = nbase + t * 16 + lane15;
    const float wn = norm_w[n];
    #pragma unroll
    for (int r = 0; r < 8; ++r) {
      const int m = r + hi * 8;
      const size_t idx = (row0 + m) * H + n;
      const float rn = acc[t][r];
      const float y  = rn * smul[m] * wn;
      if (last) {
        y_out[idx] = y;
      } else {
        resid_out[idx] = rn;
        yq_next[idx]   = f32_to_e4m3(y * inv_a);
      }
    }
  }
}

extern "C" void kernel_launch(void* const* d_in, const int* in_sizes, int n_in,
                              void* d_out, int out_size, void* d_ws, size_t ws_size,
                              hipStream_t stream) {
  const float* hidden = (const float*)d_in[0];
  const float* norm_w = (const float*)d_in[1];
  const float* lin_w  = (const float*)d_in[2];
  const float* act_s  = (const float*)d_in[3];
  const float* wt_s   = (const float*)d_in[4];
  float* out = (float*)d_out;

  unsigned char* ws  = (unsigned char*)d_ws;
  unsigned char* Wq  = ws;
  unsigned char* yqA = ws + ((size_t)4  << 20);
  unsigned char* yqB = ws + ((size_t)40 << 20);

  {
    int total4 = 3 * H * H / 4;
    quant_weights_kernel<<<(total4 + 255) / 256, 256, 0, stream>>>(
        lin_w, wt_s, Wq, total4);
  }

  relu_rms_quant_kernel<<<T_ROWS, 256, 0, stream>>>(
      hidden, norm_w, act_s, out, yqA);

  dim3 g(T_ROWS / 16), b(512);
  fp8_gemm_rms_kernel<<<g, b, 0, stream>>>(
      yqA, Wq + 0 * H * H, out, out, norm_w + 1 * H, act_s, wt_s,
      0, 0, yqB, out);
  fp8_gemm_rms_kernel<<<g, b, 0, stream>>>(
      yqB, Wq + 1 * H * H, out, out, norm_w + 2 * H, act_s, wt_s,
      1, 0, yqA, out);
  fp8_gemm_rms_kernel<<<g, b, 0, stream>>>(
      yqA, Wq + 2 * H * H, out, out, norm_w + 3 * H, act_s, wt_s,
      2, 1, nullptr, out);
}
